// DecoderBlock_25872882991393
// MI455X (gfx1250) — matrix operand order repaired and checked by reading, not compiled, not run on hardware
//
#include <hip/hip_runtime.h>
#include <hip/hip_bf16.h>

// ---------------- problem dims (fixed by reference) ----------------
#define BB       2
#define LL       2048
#define DMODEL   1024
#define DINNER   2048          // 2*DMODEL
#define DCONV    4
#define NSTATE   16
#define DTRANK   64
#define XPN      (DTRANK + 2 * NSTATE)   // 96
#define ROWS     (BB * LL)               // 4096

typedef __attribute__((ext_vector_type(16))) __bf16        v16bf;
typedef __attribute__((ext_vector_type(8)))  float         v8f;
typedef __attribute__((ext_vector_type(4)))  unsigned int  v4u;

union Frag16 { v16bf v; v4u q[2]; };

__device__ __forceinline__ unsigned short f2bf(float f) {
    unsigned int u = __float_as_uint(f);
    unsigned int r = u + 0x7fffu + ((u >> 16) & 1u);   // round-to-nearest-even
    return (unsigned short)(r >> 16);
}

__device__ __forceinline__ float silu(float x) {
    return x / (1.0f + expf(-x));
}

// ---------------- 1) RMSNorm -> bf16 rows ----------------
__global__ void k_rmsnorm(const float* __restrict__ x,
                          const float* __restrict__ w,
                          unsigned short* __restrict__ xn) {
    __shared__ float red[256];
    const int row = blockIdx.x;
    const int tid = threadIdx.x;
    const float* xr = x + (size_t)row * DMODEL;
    float s = 0.f;
    #pragma unroll
    for (int i = 0; i < 4; ++i) {
        float v = xr[tid * 4 + i];
        s += v * v;
    }
    red[tid] = s;
    __syncthreads();
    for (int ofs = 128; ofs > 0; ofs >>= 1) {
        if (tid < ofs) red[tid] += red[tid + ofs];
        __syncthreads();
    }
    float rs = rsqrtf(red[0] * (1.0f / DMODEL) + 1e-5f);
    unsigned short* orow = xn + (size_t)row * DMODEL;
    #pragma unroll
    for (int i = 0; i < 4; ++i) {
        int c = tid * 4 + i;
        orow[c] = f2bf(xr[c] * rs * w[c]);
    }
}

// ---------------- 2) transpose + fp32->bf16 : Wt[n*K+k] = W[k*N+n] ----------------
__global__ void k_transpose_bf16(const float* __restrict__ W,
                                 unsigned short* __restrict__ Wt,
                                 int K, int N) {
    long long idx = (long long)blockIdx.x * blockDim.x + threadIdx.x;
    if (idx >= (long long)K * N) return;
    int n = (int)(idx / K);
    int k = (int)(idx % K);
    Wt[idx] = f2bf(W[(size_t)k * N + n]);
}

// ---------------- 3) bf16 WMMA GEMM: C[M,N] = A[M,K] * Wt[N,K]^T ----------------
// block = 256 threads (8 waves), tile 128x64; each wave: 16 rows x 64 cols,
// 4 accumulators sharing one A fragment per K-step (2.5 vmem loads per WMMA).
__global__ void k_wmma_gemm(const unsigned short* __restrict__ A,
                            const unsigned short* __restrict__ Wt,
                            float* __restrict__ C,
                            int M, int N, int K) {
    const int wave = threadIdx.x >> 5;   // 0..7 -> M sub-tile
    const int lane = threadIdx.x & 31;
    const int h    = lane >> 4;          // lane half
    const int l15  = lane & 15;

    const int rowBase = blockIdx.y * 128 + wave * 16;
    const int colBase = blockIdx.x * 64;

    v8f c0 = {}; v8f c1 = {}; v8f c2 = {}; v8f c3 = {};

    const unsigned short* arow  = A  + (size_t)(rowBase + l15) * K;
    const unsigned short* b0row = Wt + (size_t)(colBase +  0 + l15) * K;
    const unsigned short* b1row = Wt + (size_t)(colBase + 16 + l15) * K;
    const unsigned short* b2row = Wt + (size_t)(colBase + 32 + l15) * K;
    const unsigned short* b3row = Wt + (size_t)(colBase + 48 + l15) * K;

    for (int kk = 0; kk < K; kk += 32) {
        Frag16 a, b0, b1, b2, b3;
        // A fragment: contiguous K per lane half (K = 16h .. 16h+15), matching B
        a.q[0] = *reinterpret_cast<const v4u*>(arow + kk + 16 * h);       // K = 16h .. 16h+7
        a.q[1] = *reinterpret_cast<const v4u*>(arow + kk + 16 * h + 8);   // K = 16h+8 ..
        // B fragments: lane = N, contiguous K (Wt is pre-transposed)
        {
            const v4u* p = reinterpret_cast<const v4u*>(b0row + kk + 16 * h);
            b0.q[0] = p[0]; b0.q[1] = p[1];
        }
        {
            const v4u* p = reinterpret_cast<const v4u*>(b1row + kk + 16 * h);
            b1.q[0] = p[0]; b1.q[1] = p[1];
        }
        {
            const v4u* p = reinterpret_cast<const v4u*>(b2row + kk + 16 * h);
            b2.q[0] = p[0]; b2.q[1] = p[1];
        }
        {
            const v4u* p = reinterpret_cast<const v4u*>(b3row + kk + 16 * h);
            b3.q[0] = p[0]; b3.q[1] = p[1];
        }
        c0 = __builtin_amdgcn_wmma_f32_16x16x32_bf16(false, a.v, false, b0.v,
                                                     (short)0, c0, false, false);
        c1 = __builtin_amdgcn_wmma_f32_16x16x32_bf16(false, a.v, false, b1.v,
                                                     (short)0, c1, false, false);
        c2 = __builtin_amdgcn_wmma_f32_16x16x32_bf16(false, a.v, false, b2.v,
                                                     (short)0, c2, false, false);
        c3 = __builtin_amdgcn_wmma_f32_16x16x32_bf16(false, a.v, false, b3.v,
                                                     (short)0, c3, false, false);
    }

    #pragma unroll
    for (int r = 0; r < 8; ++r) {
        int row = rowBase + 8 * h + r;
        float* crow = C + (size_t)row * N + colBase + l15;
        crow[0]  = c0[r];
        crow[16] = c1[r];
        crow[32] = c2[r];
        crow[48] = c3[r];
    }
}

// ---------------- 4) causal depthwise conv + bias + SiLU ----------------
// reads u = first DINNER columns of xr [ROWS, 2*DINNER]
__global__ void k_conv_silu(const float* __restrict__ xr,
                            const float* __restrict__ conv_w,
                            const float* __restrict__ conv_b,
                            float* __restrict__ u) {
    long long idx = (long long)blockIdx.x * blockDim.x + threadIdx.x;
    if (idx >= (long long)ROWS * DINNER) return;
    int d  = (int)(idx & (DINNER - 1));
    int rl = (int)(idx >> 11);              // row = b*L + l
    int l  = rl & (LL - 1);
    float acc = conv_b[d];
    #pragma unroll
    for (int k = 0; k < DCONV; ++k) {
        int ls = l - (DCONV - 1) + k;
        if (ls >= 0)
            acc += xr[(size_t)(rl + ls - l) * (2 * DINNER) + d] * conv_w[d * DCONV + k];
    }
    u[idx] = silu(acc);
}

// ---------------- 5) x_proj: x_dbl[row, j<96] = u_row . x_proj_w[:, j] ----------------
__global__ void k_xproj(const float* __restrict__ u,
                        const float* __restrict__ xw,
                        float* __restrict__ xdbl) {
    __shared__ float su[DINNER];
    const int row = blockIdx.x;
    const int tid = threadIdx.x;           // 128 threads
    const float* urow = u + (size_t)row * DINNER;
    for (int i = tid; i < DINNER; i += 128) su[i] = urow[i];
    __syncthreads();
    if (tid < XPN) {
        float acc = 0.f;
        for (int k = 0; k < DINNER; ++k)
            acc += su[k] * xw[(size_t)k * XPN + tid];
        xdbl[(size_t)row * XPN + tid] = acc;
    }
}

// ---------------- 6) dt_proj + softplus ----------------
__global__ void k_dtproj(const float* __restrict__ xdbl,
                         const float* __restrict__ dtw,
                         const float* __restrict__ dtb,
                         float* __restrict__ delta) {
    long long idx = (long long)blockIdx.x * blockDim.x + threadIdx.x;
    if (idx >= (long long)ROWS * DINNER) return;
    int d   = (int)(idx & (DINNER - 1));
    int row = (int)(idx >> 11);
    const float* xr = xdbl + (size_t)row * XPN;
    float acc = dtb[d];
    #pragma unroll 8
    for (int r = 0; r < DTRANK; ++r)
        acc += xr[r] * dtw[(size_t)r * DINNER + d];
    float sp = (acc > 20.f) ? acc : log1pf(expf(acc));
    delta[idx] = sp;
}

// ---------------- 7) selective scan: lane-per-state, 16-lane reduction ----------------
__global__ void k_scan(const float* __restrict__ delta,
                       const float* __restrict__ u,
                       const float* __restrict__ xdbl,
                       const float* __restrict__ A_log,
                       const float* __restrict__ D_param,
                       float* __restrict__ y) {
    int gid = blockIdx.x * blockDim.x + threadIdx.x;   // B*DINNER*16 threads
    int n  = gid & (NSTATE - 1);
    int ch = gid >> 4;                                 // b*DINNER + d
    int b  = ch >> 11;
    int d  = ch & (DINNER - 1);

    const float Adn = -expf(A_log[d * NSTATE + n]);
    const float Dd  = D_param[d];
    float h = 0.f;

    for (int l = 0; l < LL; ++l) {
        int row = b * LL + l;
        float dl = delta[(size_t)row * DINNER + d];
        float uv = u[(size_t)row * DINNER + d];
        float Bv = xdbl[(size_t)row * XPN + DTRANK + n];
        float Cv = xdbl[(size_t)row * XPN + DTRANK + NSTATE + n];
        float dA = expf(dl * Adn);
        h = dA * h + (dl * uv) * Bv;
        float p = h * Cv;
        p += __shfl_xor(p, 1);
        p += __shfl_xor(p, 2);
        p += __shfl_xor(p, 4);
        p += __shfl_xor(p, 8);
        if (n == 0) y[(size_t)row * DINNER + d] = p + uv * Dd;
    }
}

// ---------------- 8) gate: yg = bf16( y * silu(res) ) ----------------
__global__ void k_gate(const float* __restrict__ y,
                       const float* __restrict__ xr,
                       unsigned short* __restrict__ yg) {
    long long idx = (long long)blockIdx.x * blockDim.x + threadIdx.x;
    if (idx >= (long long)ROWS * DINNER) return;
    int d   = (int)(idx & (DINNER - 1));
    int row = (int)(idx >> 11);
    float res = xr[(size_t)row * (2 * DINNER) + DINNER + d];
    yg[idx] = f2bf(y[idx] * silu(res));
}

// ---------------- launch ----------------
extern "C" void kernel_launch(void* const* d_in, const int* in_sizes, int n_in,
                              void* d_out, int out_size, void* d_ws, size_t ws_size,
                              hipStream_t stream) {
    const float* x         = (const float*)d_in[0];
    const float* norm_w    = (const float*)d_in[1];
    const float* in_proj_w = (const float*)d_in[2];
    const float* conv_w    = (const float*)d_in[3];
    const float* conv_b    = (const float*)d_in[4];
    const float* x_proj_w  = (const float*)d_in[5];
    const float* dt_proj_w = (const float*)d_in[6];
    const float* dt_proj_b = (const float*)d_in[7];
    const float* A_log     = (const float*)d_in[8];
    const float* D_param   = (const float*)d_in[9];
    const float* out_proj_w= (const float*)d_in[10];
    float* out = (float*)d_out;

    char* ws = (char*)d_ws;
    size_t off = 0;
    auto alloc = [&](size_t bytes) -> void* {
        void* p = ws + off;
        off = (off + bytes + 255) & ~(size_t)255;
        return p;
    };

    unsigned short* XN    = (unsigned short*)alloc((size_t)ROWS * DMODEL * 2);
    unsigned short* WINT  = (unsigned short*)alloc((size_t)DMODEL * 2 * DINNER * 2);
    unsigned short* WOUTT = (unsigned short*)alloc((size_t)DINNER * DMODEL * 2);
    float* XR    = (float*)alloc((size_t)ROWS * 2 * DINNER * 4);
    float* U     = (float*)alloc((size_t)ROWS * DINNER * 4);
    float* XDBL  = (float*)alloc((size_t)ROWS * XPN * 4);
    float* DELTA = (float*)alloc((size_t)ROWS * DINNER * 4);
    float* Y     = (float*)alloc((size_t)ROWS * DINNER * 4);
    unsigned short* YG = (unsigned short*)alloc((size_t)ROWS * DINNER * 2);

    // 1) RMSNorm -> bf16
    k_rmsnorm<<<ROWS, 256, 0, stream>>>(x, norm_w, XN);

    // 2) weight transpose+convert (per launch; deterministic)
    {
        long long e = (long long)DMODEL * 2 * DINNER;
        k_transpose_bf16<<<(unsigned)((e + 255) / 256), 256, 0, stream>>>(in_proj_w, WINT, DMODEL, 2 * DINNER);
    }
    {
        long long e = (long long)DINNER * DMODEL;
        k_transpose_bf16<<<(unsigned)((e + 255) / 256), 256, 0, stream>>>(out_proj_w, WOUTT, DINNER, DMODEL);
    }

    // 3) in_proj GEMM: [4096,1024] x [1024,4096] -> XR
    k_wmma_gemm<<<dim3((2 * DINNER) / 64, ROWS / 128), 256, 0, stream>>>(
        XN, WINT, XR, ROWS, 2 * DINNER, DMODEL);

    // 4) conv + SiLU
    k_conv_silu<<<(ROWS * DINNER) / 256, 256, 0, stream>>>(XR, conv_w, conv_b, U);

    // 5) x_proj
    k_xproj<<<ROWS, 128, 0, stream>>>(U, x_proj_w, XDBL);

    // 6) dt_proj + softplus
    k_dtproj<<<(ROWS * DINNER) / 256, 256, 0, stream>>>(XDBL, dt_proj_w, dt_proj_b, DELTA);

    // 7) selective scan
    k_scan<<<(BB * DINNER * NSTATE) / 256, 256, 0, stream>>>(DELTA, U, XDBL, A_log, D_param, Y);

    // 8) gate -> bf16
    k_gate<<<(ROWS * DINNER) / 256, 256, 0, stream>>>(Y, XR, YG);

    // 9) out_proj GEMM: [4096,2048] x [2048,1024] -> out
    k_wmma_gemm<<<dim3(DMODEL / 64, ROWS / 128), 256, 0, stream>>>(
        YG, WOUTT, out, ROWS, DMODEL, DINNER);
}
